// Bottleneck_72739566125669
// MI455X (gfx1250) — matrix operand order repaired and checked by reading, not compiled, not run on hardware
//
#include <hip/hip_runtime.h>

// CDNA5 / gfx1250: wave32, WMMA f32_16x16x32_f16
typedef __attribute__((ext_vector_type(16))) _Float16 v16h;
typedef __attribute__((ext_vector_type(8)))  float    v8f;
typedef __attribute__((ext_vector_type(4)))  int      v4i;

#define HW   3136     // 56*56
#define NB   16       // batch
#define HH   56
#define WW   56

#if defined(__HIP_DEVICE_COMPILE__) && \
    __has_builtin(__builtin_amdgcn_global_load_async_to_lds_b128) && \
    __has_builtin(__builtin_amdgcn_s_wait_asynccnt)
#define USE_ASYNC_LDS 1
typedef __attribute__((address_space(1))) v4i* gv4i_p;   // global
typedef __attribute__((address_space(3))) v4i* lv4i_p;   // LDS
#else
#define USE_ASYNC_LDS 0
#endif

// ---------------- helpers ----------------
__device__ __forceinline__ float qstep(unsigned bits) {
  // delta = max(absmax/127, 1e-8)
  return fmaxf(__uint_as_float(bits) * (1.0f / 127.0f), 1e-8f);
}
__device__ __forceinline__ float fakequant(float v, float d) {
  return fminf(fmaxf(rintf(v / d), -127.0f), 127.0f) * d;
}

// ---------------- zero init for absmax accumulators ----------------
__global__ void zero_u32(unsigned* p, int n) {
  int i = blockIdx.x * 256 + threadIdx.x;
  if (i < n) p[i] = 0;
}

// ---------------- weight prep: fold BN, per-O-channel int8 fake quant, f16 store
// src layout OIHW flattened: w[o*K + i], i over (c,kh,kw). For 3x3 (KHW==9) the
// destination is reordered to [o][t*C + c] (tap-major, channels contiguous).
__global__ void wprep(const float* __restrict__ w, const float* __restrict__ g,
                      const float* __restrict__ b, const float* __restrict__ mu,
                      const float* __restrict__ var,
                      _Float16* __restrict__ wq, float* __restrict__ bias,
                      int C, int KHW) {
  __shared__ float red[256];
  int o = blockIdx.x;
  int K = C * KHW;
  float fact = g[o] / sqrtf(var[o] + 1e-5f);
  float lm = 0.0f;
  for (int i = threadIdx.x; i < K; i += 256)
    lm = fmaxf(lm, fabsf(w[o * K + i] * fact));
  red[threadIdx.x] = lm;
  __syncthreads();
  for (int st = 128; st > 0; st >>= 1) {
    if ((int)threadIdx.x < st) red[threadIdx.x] = fmaxf(red[threadIdx.x], red[threadIdx.x + st]);
    __syncthreads();
  }
  float delta = fmaxf(red[0] * (1.0f / 127.0f), 1e-8f);
  for (int i = threadIdx.x; i < K; i += 256) {
    float v = w[o * K + i] * fact;
    float q = fminf(fmaxf(rintf(v / delta), -127.0f), 127.0f) * delta;
    int di = (KHW == 9) ? ((i % 9) * C + i / 9) : i;   // i = c*9 + t -> t*C + c
    wq[o * K + di] = (_Float16)q;
  }
  if (threadIdx.x == 0) bias[o] = b[o] - mu[o] * fact;
}

// ---------------- conv1: 1x1, B read straight from f32 NCHW x ----------------
// W: [64][256] f16 row-major. out: f32 [16][64][3136]. amax: per out-channel bits.
__global__ void conv1x1_from_x(const _Float16* __restrict__ W,
                               const float* __restrict__ x,
                               const float* __restrict__ bias,
                               float* __restrict__ out,
                               unsigned* __restrict__ amax) {
  const int M = 64, K = 256;
  int wave = blockIdx.x * (blockDim.x >> 5) + ((int)threadIdx.x >> 5);
  int lane = threadIdx.x & 31;
  const int ngroups = (NB * HW) / 64;  // 784
  int mtile = wave / ngroups, ng = wave % ngroups;
  if (mtile >= M / 16) return;
  int n0 = ng * 64;
  int m = lane & 15, g = lane >> 4;

  int nn[4], hw[4];
  for (int s = 0; s < 4; s++) { int p = n0 + s * 16 + m; nn[s] = p / HW; hw[s] = p % HW; }

  v8f acc[4] = {};
  const _Float16* wr = W + (mtile * 16 + m) * K;
  for (int k0 = 0; k0 < K; k0 += 32) {
    v16h a;
    ((float4*)&a)[0] = *(const float4*)(wr + k0 + 16 * g);       // K = k0+16g .. +7
    ((float4*)&a)[1] = *(const float4*)(wr + k0 + 16 * g + 8);   // K = k0+16g+8 .. +15
    int cbase = k0 + 16 * g;
    for (int s = 0; s < 4; s++) {
      v16h b;
      _Float16* bh = (_Float16*)&b;
      const float* xp = x + (nn[s] * 256 + cbase) * HW + hw[s];
#pragma unroll
      for (int i = 0; i < 16; i++) bh[i] = (_Float16)xp[i * HW];  // B[k][n], n = lane&15
      acc[s] = __builtin_amdgcn_wmma_f32_16x16x32_f16(false, a, false, b,
                                                      (short)0, acc[s], false, false);
    }
  }
  float lm[8] = {};
  for (int s = 0; s < 4; s++)
    for (int r = 0; r < 8; r++) {
      int mm = mtile * 16 + r + 8 * g;
      float v = acc[s][r] + bias[mm];
      out[(nn[s] * M + mm) * HW + hw[s]] = v;
      lm[r] = fmaxf(lm[r], fabsf(v));
    }
  for (int r = 0; r < 8; r++) {
    float v = lm[r];
    for (int off = 1; off < 16; off <<= 1) v = fmaxf(v, __shfl_xor(v, off, 32));
    if ((lane & 15) == 0)
      atomicMax(&amax[mtile * 16 + r + 8 * g], __float_as_uint(v));
  }
}

// ---------------- quantize raw f32 NCHW -> channels-last f16 (optional ReLU) ----
// idx runs over the channels-last destination so the f16 stores are coalesced;
// the strided f32 reads hit L2 (raw buffer was just produced, 12.8 MB << 192 MB).
__global__ void quant_to_cl(const float* __restrict__ raw, const unsigned* __restrict__ amax,
                            _Float16* __restrict__ act, int C, int relu) {
  int idx = blockIdx.x * 256 + threadIdx.x;
  if (idx >= NB * C * HW) return;
  int c = idx % C; int t = idx / C; int hw = t % HW; int nn = t / HW;
  float q = fakequant(raw[(nn * C + c) * HW + hw], qstep(amax[c]));
  if (relu) q = fmaxf(q, 0.0f);
  act[idx] = (_Float16)q;
}

// ---------------- conv2: 3x3 pad 1, im2col-on-the-fly from channels-last f16 ----
// W: [64][9*64] f16 (tap-major). A: [50176][64] f16. out: f32 [16][64][3136].
__global__ void conv3x3_wmma(const _Float16* __restrict__ W,
                             const _Float16* __restrict__ A,
                             const float* __restrict__ bias,
                             float* __restrict__ out,
                             unsigned* __restrict__ amax) {
  const int M = 64;
  int wave = blockIdx.x * (blockDim.x >> 5) + ((int)threadIdx.x >> 5);
  int lane = threadIdx.x & 31;
  const int ngroups = (NB * HW) / 64;
  int mtile = wave / ngroups, ng = wave % ngroups;
  if (mtile >= 4) return;
  int n0 = ng * 64;
  int m = lane & 15, g = lane >> 4;

  int nn[4], h[4], w[4];
  for (int s = 0; s < 4; s++) {
    int p = n0 + s * 16 + m; nn[s] = p / HW; int hw = p % HW; h[s] = hw / WW; w[s] = hw % WW;
  }

  v8f acc[4] = {};
  const _Float16* wr = W + (mtile * 16 + m) * 576;
  for (int t = 0; t < 9; t++) {
    int dh = t / 3 - 1, dw = t % 3 - 1;
    for (int kc = 0; kc < 64; kc += 32) {
      v16h a;
      ((float4*)&a)[0] = *(const float4*)(wr + t * 64 + kc + 16 * g);
      ((float4*)&a)[1] = *(const float4*)(wr + t * 64 + kc + 16 * g + 8);
      for (int s = 0; s < 4; s++) {
        v16h b = {};
        int h2 = h[s] + dh, w2 = w[s] + dw;
        if ((unsigned)h2 < (unsigned)HH && (unsigned)w2 < (unsigned)WW) {
          const _Float16* br = A + ((nn[s] * HW) + h2 * WW + w2) * 64 + kc + 16 * g;
          ((float4*)&b)[0] = *(const float4*)(br);
          ((float4*)&b)[1] = *(const float4*)(br + 8);
        }
        acc[s] = __builtin_amdgcn_wmma_f32_16x16x32_f16(false, a, false, b,
                                                        (short)0, acc[s], false, false);
      }
    }
  }
  float lm[8] = {};
  for (int s = 0; s < 4; s++) {
    int hw = h[s] * WW + w[s];
    for (int r = 0; r < 8; r++) {
      int mm = mtile * 16 + r + 8 * g;
      float v = acc[s][r] + bias[mm];
      out[(nn[s] * M + mm) * HW + hw] = v;
      lm[r] = fmaxf(lm[r], fabsf(v));
    }
  }
  for (int r = 0; r < 8; r++) {
    float v = lm[r];
    for (int off = 1; off < 16; off <<= 1) v = fmaxf(v, __shfl_xor(v, off, 32));
    if ((lane & 15) == 0)
      atomicMax(&amax[mtile * 16 + r + 8 * g], __float_as_uint(v));
  }
}

// ---------------- conv3: 1x1 GEMM 64 -> 256, block-cooperative LDS B-tile -------
// Block = 8 waves. The 64-position x 64-K B tile (8 KB) is staged into LDS ONCE
// (async copy on gfx1250), then shared by 8 M-tiles -> 8x less B read traffic.
// LDS rows padded to 72 halves so per-lane ds_load_b128 is bank-conflict-free.
__global__ void conv1x1_cl(const _Float16* __restrict__ W,   // [256][64]
                           const _Float16* __restrict__ A,   // [50176][64]
                           const float* __restrict__ bias,
                           float* __restrict__ out,          // [16][256][3136]
                           unsigned* __restrict__ amax) {
  const int M = 256, K = 64;
  __shared__ _Float16 Bs[64 * 72];

  int mhalf = blockIdx.x & 1;
  int ng    = blockIdx.x >> 1;
  int n0    = ng * 64;
  int wavei = (int)threadIdx.x >> 5;
  int lane  = threadIdx.x & 31;
  int mtile = mhalf * 8 + wavei;

  // ---- stage B tile: 64 rows x 128 B = 512 x 16 B chunks, 2 per thread ----
  for (int ch = threadIdx.x; ch < 512; ch += 256) {
    int row = ch >> 3, o16 = ch & 7;
    const _Float16* src = A + (n0 + row) * K + o16 * 8;
    _Float16* dst = Bs + row * 72 + o16 * 8;
#if USE_ASYNC_LDS
    __builtin_amdgcn_global_load_async_to_lds_b128((gv4i_p)src, (lv4i_p)dst, 0, 0);
#else
    *(float4*)dst = *(const float4*)src;
#endif
  }
#if USE_ASYNC_LDS
  __builtin_amdgcn_s_wait_asynccnt(0);
#endif
  __syncthreads();

  int m = lane & 15, g = lane >> 4;
  int nn[4], hw[4];
  for (int s = 0; s < 4; s++) { int p = n0 + s * 16 + m; nn[s] = p / HW; hw[s] = p % HW; }

  v8f acc[4] = {};
  const _Float16* wr = W + (mtile * 16 + m) * K;
  for (int k0 = 0; k0 < K; k0 += 32) {
    v16h a;
    ((float4*)&a)[0] = *(const float4*)(wr + k0 + 16 * g);
    ((float4*)&a)[1] = *(const float4*)(wr + k0 + 16 * g + 8);
    for (int s = 0; s < 4; s++) {
      v16h b;
      const _Float16* br = Bs + (s * 16 + m) * 72 + k0 + 16 * g;  // column n = lane&15
      ((float4*)&b)[0] = *(const float4*)(br);
      ((float4*)&b)[1] = *(const float4*)(br + 8);
      acc[s] = __builtin_amdgcn_wmma_f32_16x16x32_f16(false, a, false, b,
                                                      (short)0, acc[s], false, false);
    }
  }
  float lm[8] = {};
  for (int s = 0; s < 4; s++)
    for (int r = 0; r < 8; r++) {
      int mm = mtile * 16 + r + 8 * g;
      float v = acc[s][r] + bias[mm];
      out[(nn[s] * M + mm) * HW + hw[s]] = v;
      lm[r] = fmaxf(lm[r], fabsf(v));
    }
  for (int r = 0; r < 8; r++) {
    float v = lm[r];
    for (int off = 1; off < 16; off <<= 1) v = fmaxf(v, __shfl_xor(v, off, 32));
    if ((lane & 15) == 0)
      atomicMax(&amax[mtile * 16 + r + 8 * g], __float_as_uint(v));
  }
}

// ---------------- residual: pass 1, per-channel absmax of quant(raw3)+x --------
__global__ void resid_absmax(const float* __restrict__ raw3, const float* __restrict__ x,
                             const unsigned* __restrict__ amax3, unsigned* __restrict__ amaxs) {
  __shared__ float red[256];
  int c = blockIdx.x & 255;                 // block = (nn*256 + c)
  int base = blockIdx.x * HW;
  float d3 = qstep(amax3[c]);
  float lm = 0.0f;
  for (int i = threadIdx.x; i < HW; i += 256) {
    float s = fakequant(raw3[base + i], d3) + x[base + i];
    lm = fmaxf(lm, fabsf(s));
  }
  red[threadIdx.x] = lm;
  __syncthreads();
  for (int st = 128; st > 0; st >>= 1) {
    if ((int)threadIdx.x < st) red[threadIdx.x] = fmaxf(red[threadIdx.x], red[threadIdx.x + st]);
    __syncthreads();
  }
  if (threadIdx.x == 0) atomicMax(&amaxs[c], __float_as_uint(red[0]));
}

// ---------------- residual: pass 2, recompute + final quant + ReLU -------------
__global__ void resid_final(const float* __restrict__ raw3, const float* __restrict__ x,
                            const unsigned* __restrict__ amax3, const unsigned* __restrict__ amaxs,
                            float* __restrict__ out) {
  int idx = blockIdx.x * 256 + threadIdx.x;
  if (idx >= NB * 256 * HW) return;
  int c = (idx / HW) & 255;
  float s = fakequant(raw3[idx], qstep(amax3[c])) + x[idx];
  float q = fakequant(s, qstep(amaxs[c]));
  out[idx] = fmaxf(q, 0.0f);
}

// ---------------- host launcher ----------------
extern "C" void kernel_launch(void* const* d_in, const int* in_sizes, int n_in,
                              void* d_out, int out_size, void* d_ws, size_t ws_size,
                              hipStream_t stream) {
  (void)in_sizes; (void)n_in; (void)out_size; (void)ws_size;
  const float* x  = (const float*)d_in[0];
  const float* w1 = (const float*)d_in[1];
  const float* g1 = (const float*)d_in[2];
  const float* b1 = (const float*)d_in[3];
  const float* m1 = (const float*)d_in[4];
  const float* v1 = (const float*)d_in[5];
  const float* w2 = (const float*)d_in[6];
  const float* g2 = (const float*)d_in[7];
  const float* b2 = (const float*)d_in[8];
  const float* m2 = (const float*)d_in[9];
  const float* v2 = (const float*)d_in[10];
  const float* w3 = (const float*)d_in[11];
  const float* g3 = (const float*)d_in[12];
  const float* b3 = (const float*)d_in[13];
  const float* m3 = (const float*)d_in[14];
  const float* v3 = (const float*)d_in[15];

  char* ws = (char*)d_ws;
  size_t off = 0;
  auto alloc = [&](size_t bytes) -> void* {
    void* p = ws + off;
    off = (off + bytes + 255) & ~(size_t)255;
    return p;
  };
  _Float16* wq1   = (_Float16*)alloc((size_t)64 * 256 * 2);
  _Float16* wq2   = (_Float16*)alloc((size_t)64 * 576 * 2);
  _Float16* wq3   = (_Float16*)alloc((size_t)256 * 64 * 2);
  float*    bias1 = (float*)alloc(64 * 4);
  float*    bias2 = (float*)alloc(64 * 4);
  float*    bias3 = (float*)alloc(256 * 4);
  unsigned* amaxA = (unsigned*)alloc(640 * 4);     // [64 | 64 | 256 | 256]
  unsigned* amax1 = amaxA;
  unsigned* amax2 = amaxA + 64;
  unsigned* amax3 = amaxA + 128;
  unsigned* amaxs = amaxA + 384;
  float*    raw1  = (float*)alloc((size_t)NB * 64 * HW * 4);    // reused as raw2
  _Float16* a1    = (_Float16*)alloc((size_t)NB * HW * 64 * 2); // reused as a2
  float*    raw3  = (float*)alloc((size_t)NB * 256 * HW * 4);
  float*    raw2  = raw1;   // raw1 dead after quant1
  _Float16* a2    = a1;     // a1 dead after conv2

  // absmax accumulators must start at 0 every call
  zero_u32<<<3, 256, 0, stream>>>(amaxA, 640);

  // fold BN + per-channel int8 fake-quant weights -> f16
  wprep<<<64, 256, 0, stream>>>(w1, g1, b1, m1, v1, wq1, bias1, 256, 1);
  wprep<<<64, 256, 0, stream>>>(w2, g2, b2, m2, v2, wq2, bias2, 64, 9);
  wprep<<<256, 256, 0, stream>>>(w3, g3, b3, m3, v3, wq3, bias3, 64, 1);

  // stage 1: 1x1 conv from raw f32 x
  conv1x1_from_x<<<392, 256, 0, stream>>>(wq1, x, bias1, raw1, amax1);
  quant_to_cl<<<12544, 256, 0, stream>>>(raw1, amax1, a1, 64, 1);

  // stage 2: 3x3 conv
  conv3x3_wmma<<<392, 256, 0, stream>>>(wq2, a1, bias2, raw2, amax2);
  quant_to_cl<<<12544, 256, 0, stream>>>(raw2, amax2, a2, 64, 1);

  // stage 3: 1x1 conv 64 -> 256 (LDS-cooperative, async staging)
  conv1x1_cl<<<1568, 256, 0, stream>>>(wq3, a2, bias3, raw3, amax3);

  // residual add + final per-channel quant + ReLU
  resid_absmax<<<4096, 256, 0, stream>>>(raw3, x, amax3, amaxs);
  resid_final<<<50176, 256, 0, stream>>>(raw3, x, amax3, amaxs, (float*)d_out);
}
